// TOTEMVQVAE_34961033790150
// MI455X (gfx1250) — matrix operand order repaired and checked by reading, not compiled, not run on hardware
//
#include <hip/hip_runtime.h>
#include <hip/hip_bf16.h>
#include <stddef.h>

// ---------------- problem constants ----------------
#define IN_CH   8
#define L_IN    262144
#define LAT     1024
#define NEMB    1024
#define L2LEN   131072     // encoder output length (stride 2, SAME)
#define KCONV   4

typedef __attribute__((ext_vector_type(16))) __bf16 v16bf;
typedef __attribute__((ext_vector_type(8)))  __bf16 v8bf;
typedef __attribute__((ext_vector_type(8)))  float  v8f;

__device__ __forceinline__ __bf16 f2bf(float f) {
    unsigned u = __builtin_bit_cast(unsigned, f);
    unsigned r = (u + 0x7FFFu + ((u >> 16) & 1u)) >> 16;   // round-to-nearest-even
    unsigned short h = (unsigned short)r;
    return __builtin_bit_cast(__bf16, h);
}

// ============================================================================
// 1) Embedding prep: bf16 copy of E (NEMB x LAT, row major) + 0.5*||e_n||^2
// ============================================================================
__global__ __launch_bounds__(256)
void prep_emb_kernel(const float* __restrict__ emb,
                     __bf16* __restrict__ Ebf,
                     float* __restrict__ en2h) {
    __shared__ float red[256];
    const int n = blockIdx.x;
    float s = 0.f;
    for (int i = threadIdx.x; i < LAT; i += 256) {
        float v = emb[(size_t)n * LAT + i];
        s += v * v;
        Ebf[(size_t)n * LAT + i] = f2bf(v);
    }
    red[threadIdx.x] = s;
    __syncthreads();
    for (int st = 128; st > 0; st >>= 1) {
        if (threadIdx.x < st) red[threadIdx.x] += red[threadIdx.x + st];
        __syncthreads();
    }
    if (threadIdx.x == 0) en2h[n] = 0.5f * red[0];
}

// ============================================================================
// 2) Encoder: stride-2 SAME conv (pad_left=1).  z_e[c,t] f32 (coalesced over t)
//    + bf16 A-matrix in (t, LAT) row-major via LDS transpose tile.
// ============================================================================
#define ENC_T 64   // time positions per block

__global__ __launch_bounds__(256)
void encoder_kernel(const float* __restrict__ x,     // (8, L_IN)
                    const float* __restrict__ w,     // (LAT, 8, 4)
                    const float* __restrict__ b,     // (LAT)
                    float* __restrict__ ze,          // (LAT, L2LEN)
                    __bf16* __restrict__ A) {        // (L2LEN, LAT)
    __shared__ __attribute__((aligned(16))) float  lx[IN_CH][132];      // x window
    __shared__ __attribute__((aligned(16))) __bf16 lT[ENC_T][72];       // transpose tile

    const int tid = threadIdx.x;
    const int t0  = blockIdx.x * ENC_T;

    // stage x window: lx[ic][u] = x[ic, 2*t0 - 1 + u], u in [0,129] used
    for (int i = tid; i < IN_CH * 132; i += 256) {
        int ic = i / 132, u = i % 132;
        long g = (long)2 * t0 - 1 + u;
        float v = 0.f;
        if (u < 131 && g >= 0 && g < L_IN) v = x[(size_t)ic * L_IN + g];
        lx[ic][u] = v;
    }
    __syncthreads();

    const int tx = tid & 63;   // time offset within tile
    const int ty = tid >> 6;   // 0..3 channel sub-lane

    for (int c0 = 0; c0 < LAT; c0 += 64) {
        for (int cc = 0; cc < 64; cc += 4) {
            const int c = c0 + cc + ty;
            float acc = b[c];
            const float* wc = w + (size_t)c * 32;
#pragma unroll
            for (int ic = 0; ic < IN_CH; ++ic)
#pragma unroll
                for (int k = 0; k < KCONV; ++k)
                    acc += wc[ic * 4 + k] * lx[ic][2 * tx + k];
            ze[(size_t)c * L2LEN + t0 + tx] = acc;     // coalesced over tx
            lT[tx][cc + ty] = f2bf(acc);
        }
        __syncthreads();
        // dump 64 t-rows x 32 c (bf16) -> A[(t0+row), c0 + ...], 16B per thread
        {
            int row = tid >> 2, seg = tid & 3;
            uint4 v = *(const uint4*)&lT[row][seg * 8];
            *(uint4*)(A + (size_t)(t0 + row) * LAT + c0 + seg * 8) = v;
        }
        __syncthreads();
    }
}

// ============================================================================
// 3) VQ nearest-code search: bf16 WMMA GEMM  dot = zt @ E.T  with fused
//    argmax_n (dot - 0.5||e_n||^2) epilogue (== argmin distance).
//    Block: 4 waves / 128 threads as a 2(M) x 2(N) wave grid.
//    Each wave: 2x2 register blocking (32x32 C macro-tile) -> 2 ds_load/WMMA.
//    LDS: A tile 64 rows x full K = 128KB; B tile 64 rows x 256 K = 32KB
//         (quarter-K staging, accumulators carried across slices) = 160KB.
// ============================================================================
#define M_BLOCK 64
#define N_STEP  64
#define K_SLICE 256
#define K_STEP  32

__global__ __launch_bounds__(128)
void vq_argmin_gemm(const __bf16* __restrict__ A,   // (L2LEN, LAT) row-major
                    const __bf16* __restrict__ E,   // (NEMB,  LAT) row-major
                    const float* __restrict__ en2h, // 0.5*||e_n||^2
                    int* __restrict__ idx_i,
                    float* __restrict__ idx_f) {
    __shared__ __attribute__((aligned(16))) __bf16 lA[M_BLOCK * LAT];     // 128 KB
    __shared__ __attribute__((aligned(16))) __bf16 lB[N_STEP * K_SLICE];  //  32 KB

    const int tid  = threadIdx.x;
    const int wave = tid >> 5;
    const int lane = tid & 31;
    const int wm   = wave & 1;       // M half of the wave grid
    const int wn   = wave >> 1;      // N half of the wave grid
    const int m0   = blockIdx.x * M_BLOCK;

    // stage A tile (rows m0..m0+63 contiguous in global), 8192 uint4
    {
        const uint4* src = (const uint4*)(A + (size_t)m0 * LAT);
        uint4* dst = (uint4*)lA;
        for (int i = tid; i < (M_BLOCK * LAT) / 8; i += 128) dst[i] = src[i];
    }

    const int l16   = lane & 15;
    const int khalf = lane >> 4;     // 0 or 1

    // per-lane A/B row pointers for the 2x2 macro-tile
    const __bf16* aRow[2];
    const __bf16* bRow[2];
#pragma unroll
    for (int t = 0; t < 2; ++t)
        aRow[t] = lA + (size_t)(wm * 32 + t * 16 + l16) * LAT;
#pragma unroll
    for (int u = 0; u < 2; ++u)
        bRow[u] = lB + (size_t)(wn * 32 + u * 16 + l16) * K_SLICE;

    float best[2][8];
    int   bidx[2][8];
#pragma unroll
    for (int t = 0; t < 2; ++t)
#pragma unroll
        for (int r = 0; r < 8; ++r) { best[t][r] = -3.4e38f; bidx[t][r] = 0; }

    for (int n0 = 0; n0 < NEMB; n0 += N_STEP) {
        v8f acc[2][2];
#pragma unroll
        for (int t = 0; t < 2; ++t)
#pragma unroll
            for (int u = 0; u < 2; ++u) acc[t][u] = (v8f){};

        for (int ks = 0; ks < LAT; ks += K_SLICE) {
            __syncthreads();   // protect lB readers (and lA writes on first pass)
            // stage B slice: rows n0..n0+63, K cols ks..ks+255  (2048 uint4)
            {
                for (int i = tid; i < (N_STEP * K_SLICE) / 8; i += 128) {
                    int row = i >> 5, seg = i & 31;
                    *((uint4*)lB + i) =
                        *(const uint4*)(E + (size_t)(n0 + row) * LAT + ks + seg * 8);
                }
            }
            __syncthreads();

            for (int kk = 0; kk < K_SLICE; kk += K_STEP) {
                union { v8bf h[2]; v16bf v; } a[2], bmat[2];
#pragma unroll
                for (int t = 0; t < 2; ++t) {
                    // A 16-bit 16x32 layout: lanes 0-15 K {0..7,16..23}, lanes 16-31 {8..15,24..31}
                    a[t].h[0] = *(const v8bf*)(aRow[t] + ks + kk + khalf * 8);
                    a[t].h[1] = *(const v8bf*)(aRow[t] + ks + kk + 16 + khalf * 8);
                }
#pragma unroll
                for (int u = 0; u < 2; ++u) {
                    // B 32x16 layout: lanes 0-15 K {0..7,16..23}, lanes 16-31 {8..15,24..31}
                    bmat[u].h[0] = *(const v8bf*)(bRow[u] + kk + khalf * 8);
                    bmat[u].h[1] = *(const v8bf*)(bRow[u] + kk + 16 + khalf * 8);
                }
#pragma unroll
                for (int t = 0; t < 2; ++t)
#pragma unroll
                    for (int u = 0; u < 2; ++u)
                        acc[t][u] = __builtin_amdgcn_wmma_f32_16x16x32_bf16(
                            false, a[t].v, false, bmat[u].v, (short)0, acc[t][u],
                            false, false);
            }
        }

        // epilogue: C layout -> lane holds N = ntile + l16; VGPR r holds M = r + 8*khalf
#pragma unroll
        for (int u = 0; u < 2; ++u) {
            const int n   = n0 + wn * 32 + u * 16 + l16;
            const float sub = en2h[n];
#pragma unroll
            for (int t = 0; t < 2; ++t)
#pragma unroll
                for (int r = 0; r < 8; ++r) {
                    float s = acc[t][u][r] - sub;
                    if (s > best[t][r]) { best[t][r] = s; bidx[t][r] = n; }
                }
        }
    }

    // intra-wave reduce across the 16 lanes of each half (same M row, different N)
#pragma unroll
    for (int off = 8; off >= 1; off >>= 1) {
#pragma unroll
        for (int t = 0; t < 2; ++t)
#pragma unroll
            for (int r = 0; r < 8; ++r) {
                float ov = __shfl_xor(best[t][r], off, 32);
                int   oi = __shfl_xor(bidx[t][r], off, 32);
                if (ov > best[t][r] || (ov == best[t][r] && oi < bidx[t][r])) {
                    best[t][r] = ov; bidx[t][r] = oi;
                }
            }
    }

    // cross-wave (wn=0 vs wn=1) reduce via LDS scratch aliased onto lA
    __syncthreads();                 // all waves done reading lA
    float* redv = (float*)lA;        // [64 rows][2 wn]
    int*   redi = (int*)((char*)lA + 512);
    if (l16 == 0) {                  // lane 0 -> M=r, lane 16 -> M=r+8
#pragma unroll
        for (int t = 0; t < 2; ++t)
#pragma unroll
            for (int r = 0; r < 8; ++r) {
                int row = wm * 32 + t * 16 + khalf * 8 + r;
                redv[row * 2 + wn] = best[t][r];
                redi[row * 2 + wn] = bidx[t][r];
            }
    }
    __syncthreads();
    if (tid < M_BLOCK) {
        float v0 = redv[tid * 2 + 0], v1 = redv[tid * 2 + 1];
        int   i0 = redi[tid * 2 + 0], i1 = redi[tid * 2 + 1];
        int   bi = (v1 > v0 || (v1 == v0 && i1 < i0)) ? i1 : i0;
        idx_i[m0 + tid] = bi;
        idx_f[m0 + tid] = (float)bi;
    }
}

// ============================================================================
// 4) z_q gather: z_q[c,t] = emb[idx[t], c]; coalesced stores, L2-resident gathers
// ============================================================================
__global__ __launch_bounds__(256)
void zq_gather_kernel(const int* __restrict__ idx,
                      const float* __restrict__ emb,
                      float* __restrict__ zq) {
    const int t = blockIdx.x * 256 + threadIdx.x;
    const float* row = emb + (size_t)idx[t] * LAT;
    for (int c = 0; c < LAT; ++c)
        zq[(size_t)c * L2LEN + t] = row[c];
}

// ============================================================================
// 5) Decoder table: P[j,k,oc] = sum_c tconv_w[oc,c,k] * emb[j,c]
// ============================================================================
__global__ __launch_bounds__(64)
void precompute_P_kernel(const float* __restrict__ emb,
                         const float* __restrict__ tw,   // (8, LAT, 4)
                         float* __restrict__ P) {        // (NEMB, 4, 8)
    __shared__ float le[LAT];
    const int j = blockIdx.x;
    for (int i = threadIdx.x; i < LAT; i += 64) le[i] = emb[(size_t)j * LAT + i];
    __syncthreads();
    if (threadIdx.x < 32) {
        const int k = threadIdx.x >> 3, oc = threadIdx.x & 7;
        float s = 0.f;
        for (int c = 0; c < LAT; ++c)
            s += tw[((size_t)oc * LAT + c) * 4 + k] * le[c];
        P[(size_t)j * 32 + k * 8 + oc] = s;
    }
}

// ============================================================================
// 6) Decoder: x_recon[oc,t] = P[idx[p0],k0,oc] + P[idx[p1],k1,oc] + tb[oc]
//    (ConvTranspose k=4,s=2,pad(2,2) collapsed to two table rows per output)
// ============================================================================
__global__ __launch_bounds__(256)
void recon_kernel(const int* __restrict__ idx,
                  const float* __restrict__ P,
                  const float* __restrict__ tb,
                  float* __restrict__ xr) {
    const int t = blockIdx.x * 256 + threadIdx.x;
    int p0, k0, p1, k1;
    if (t & 1) { p0 = (t - 1) >> 1; k0 = 1; p1 = (t + 1) >> 1; k1 = 3; }
    else       { p0 = (t - 2) >> 1; k0 = 0; p1 = t >> 1;       k1 = 2; }
    const bool v0 = (p0 >= 0) && (p0 < L2LEN);
    const bool v1 = (p1 >= 0) && (p1 < L2LEN);
    const float* r0 = v0 ? (P + (size_t)idx[p0] * 32 + k0 * 8) : nullptr;
    const float* r1 = v1 ? (P + (size_t)idx[p1] * 32 + k1 * 8) : nullptr;
#pragma unroll
    for (int oc = 0; oc < IN_CH; ++oc) {
        float v = tb[oc];
        if (r0) v += r0[oc];
        if (r1) v += r1[oc];
        xr[(size_t)oc * L_IN + t] = v;
    }
}

// ============================================================================
extern "C" void kernel_launch(void* const* d_in, const int* in_sizes, int n_in,
                              void* d_out, int out_size, void* d_ws, size_t ws_size,
                              hipStream_t stream) {
    const float* x   = (const float*)d_in[0];   // (8, 262144)
    const float* cw  = (const float*)d_in[1];   // (1024, 8, 4)
    const float* cb  = (const float*)d_in[2];   // (1024, 1)
    const float* emb = (const float*)d_in[3];   // (1024, 1024)
    const float* tw  = (const float*)d_in[4];   // (8, 1024, 4)
    const float* tb  = (const float*)d_in[5];   // (8, 1)

    // outputs concatenated: x_recon, z_e, z_q, indices
    float* out  = (float*)d_out;
    float* xr   = out;                                   // 8*262144
    float* ze   = out + (size_t)IN_CH * L_IN;            // 1024*131072
    float* zq   = ze  + (size_t)LAT * L2LEN;             // 1024*131072
    float* idxf = zq  + (size_t)LAT * L2LEN;             // 131072

    // workspace carve-out (~259 MB)
    char* ws = (char*)d_ws;
    size_t off = 0;
    __bf16* Abf  = (__bf16*)(ws + off); off += (size_t)L2LEN * LAT * 2;  // 256 MB
    __bf16* Ebf  = (__bf16*)(ws + off); off += (size_t)NEMB * LAT * 2;   //   2 MB
    float*  en2h = (float*) (ws + off); off += (size_t)NEMB * 4;
    int*    idxi = (int*)   (ws + off); off += (size_t)L2LEN * 4;
    float*  P    = (float*) (ws + off); off += (size_t)NEMB * 32 * 4;

    prep_emb_kernel    <<<NEMB, 256, 0, stream>>>(emb, Ebf, en2h);
    encoder_kernel     <<<L2LEN / ENC_T, 256, 0, stream>>>(x, cw, cb, ze, Abf);
    precompute_P_kernel<<<NEMB, 64, 0, stream>>>(emb, tw, P);
    vq_argmin_gemm     <<<L2LEN / M_BLOCK, 128, 0, stream>>>(Abf, Ebf, en2h, idxi, idxf);
    zq_gather_kernel   <<<L2LEN / 256, 256, 0, stream>>>(idxi, emb, zq);
    recon_kernel       <<<L_IN / 256, 256, 0, stream>>>(idxi, P, tb, xr);
}
